// AlignGrapher_24807731101831
// MI455X (gfx1250) — matrix operand order repaired and checked by reading, not compiled, not run on hardware
//
#include <hip/hip_runtime.h>
#include <hip/hip_bf16.h>
#include <cstdint>

// ---------------------------------------------------------------------------
// AlignGrapher for MI455X (gfx1250): all GEMMs via v_wmma_f32_16x16x32_f16,
// strided tile feeds via the Tensor Data Mover (tensor_load_to_lds +
// s_wait_tensorcnt), double-buffered in the heavy conv GEMM.
// B=4, C=64, IMG=112, P=8, E=4096, K=9. f16 inputs / f32 accumulate.
// ---------------------------------------------------------------------------

typedef _Float16     v16h __attribute__((ext_vector_type(16)));
typedef float        v8f  __attribute__((ext_vector_type(8)));
typedef unsigned int v4u  __attribute__((ext_vector_type(4)));
typedef int          v4i  __attribute__((ext_vector_type(4)));
typedef int          v8i  __attribute__((ext_vector_type(8)));

#define NB   4
#define NC   64
#define NIMG 112
#define NHW  12544      // 112*112
#define NE   4096       // P*P*C
#define NS   256        // 16*16 conv output spatial
#define NZ   3136       // 56*56
#define NK   9
#define NTOT 3211264    // 4*64*112*112

__device__ __forceinline__ v8f v8f_splat(float x) {
  v8f r;
#pragma unroll
  for (int i = 0; i < 8; ++i) r[i] = x;
  return r;
}

__device__ __forceinline__ v8f wmma32(v16h a, v16h b, v8f c) {
  // D = A(16x32 f16) * B(32x16 f16) + C(16x16 f32)
  return __builtin_amdgcn_wmma_f32_16x16x32_f16(false, a, false, b, (short)0, c,
                                                false, false);
}

// ---------------------------------------------------------------------------
// Tensor Data Mover: 2D tile load (global -> LDS), D# per CDNA5 ISA ch.8.
// dims/strides in data_size units; data_size_code: 0=1B,1=2B,2=4B,3=8B.
// Issue from one wave; pair with s_wait_tensorcnt + workgroup barrier.
// ---------------------------------------------------------------------------
__device__ __forceinline__ void tdm_load_2d(unsigned lds_addr, const void* gptr,
                                            unsigned tile_d0, unsigned tile_d1,
                                            unsigned tensor_d0, unsigned tensor_d1,
                                            unsigned stride_d0,
                                            unsigned data_size_code) {
  unsigned long long ga = (unsigned long long)(uintptr_t)gptr;
  v4u g0;
  g0[0] = 1u;                                            // count=1, user D#
  g0[1] = lds_addr;                                      // LDS byte address
  g0[2] = (unsigned)(ga & 0xffffffffu);                  // global_addr[31:0]
  g0[3] = (unsigned)((ga >> 32) & 0x01ffffffu) | (2u << 30);  // [56:32] | type=2
  v8i g1;
  g1[0] = (int)(data_size_code << 16);                   // mask=0, data_size
  g1[1] = (int)((tensor_d0 & 0xffffu) << 16);            // tensor_dim0 lo16
  g1[2] = (int)(((tensor_d0 >> 16) & 0xffffu) | ((tensor_d1 & 0xffffu) << 16));
  g1[3] = (int)(((tensor_d1 >> 16) & 0xffffu) | ((tile_d0 & 0xffffu) << 16));
  g1[4] = (int)(tile_d1 & 0xffffu);                      // tile_dim1, tile_dim2=0
  g1[5] = (int)stride_d0;                                // dim0 stride lo32
  g1[6] = 0;                                             // stride hi16 | dim1 stride lo16
  g1[7] = 0;
  v4i gz = {0, 0, 0, 0};
#if __clang_major__ >= 23
  v8i gz8 = {0, 0, 0, 0, 0, 0, 0, 0};
  __builtin_amdgcn_tensor_load_to_lds(g0, g1, gz, gz, gz8, 0);
#else
  __builtin_amdgcn_tensor_load_to_lds(g0, g1, gz, gz, 0);
#endif
}

__device__ __forceinline__ unsigned lds_addr_of(const void* p) {
  // generic pointers to LDS carry the byte offset in the low 32 bits (ISA 10.2)
  return (unsigned)(unsigned long long)(uintptr_t)p;
}

// A fragment (16x32 f16), row-major src, leading dim ld (elements).
// Layout: lanes 0-15 hold row=lane, K=0..15; lanes 16-31 K=16..31 (matches B fragments).
__device__ __forceinline__ v16h frag_a_f16(const _Float16* src, int ld) {
  int lane = threadIdx.x & 31;
  const _Float16* p = src + (lane & 15) * ld + ((lane >> 4) * 16);
  v16h r;
#pragma unroll
  for (int j = 0; j < 8; ++j) { r[j] = p[j]; r[j + 8] = p[8 + j]; }
  return r;
}

__device__ __forceinline__ v16h frag_a_f32(const float* src, int ld) {
  int lane = threadIdx.x & 31;
  const float* p = src + (lane & 15) * ld + ((lane >> 4) * 16);
  v16h r;
#pragma unroll
  for (int j = 0; j < 8; ++j) {
    r[j]     = (_Float16)p[j];
    r[j + 8] = (_Float16)p[8 + j];
  }
  return r;
}

// B fragment (32x16 f16), column-major src (each column contiguous in K), ld = column stride.
// Layout used here (contiguous k, same as A): lanes 0-15 hold col=lane, K=0..15; lanes 16-31 hold K=16..31.
__device__ __forceinline__ v16h frag_b_f16(const _Float16* src, int ld) {
  int lane = threadIdx.x & 31;
  const _Float16* p = src + (lane & 15) * ld + ((lane >> 4) * 16);
  v16h r;
#pragma unroll
  for (int j = 0; j < 16; ++j) r[j] = p[j];
  return r;
}

// B fragment from K-major f32 tile (B[k][n], row length ld): TDM-staged tiles.
__device__ __forceinline__ v16h frag_b_f32k(const float* src, int ld) {
  int lane = threadIdx.x & 31;
  const float* p = src + (lane & 15) + ((lane >> 4) * 16) * ld;
  v16h r;
#pragma unroll
  for (int j = 0; j < 16; ++j) r[j] = (_Float16)p[j * ld];
  return r;
}

// ---------------------------------------------------------------------------
// Kernel 0: zero BN statistic accumulators (3 sets of 128 floats)
// ---------------------------------------------------------------------------
__global__ void k_zero(float* p, int n) {
  int i = blockIdx.x * blockDim.x + threadIdx.x;
  if (i < n) p[i] = 0.f;
}

// ---------------------------------------------------------------------------
// Kernel 1: c_aggregation conv as im2col GEMM. Per batch: (4096 x 4096) @ (4096 x 256).
// Grid (64, 4, B), block 128 (4 waves). WG tile M=64, N=64; wave = 16 M-rows x 64 N-cols.
// A tiles: double-buffered TDM (f32 -> LDS). B tiles: branch-free fused im2col.
// ---------------------------------------------------------------------------
__global__ __launch_bounds__(128) void k_cagg(const float* __restrict__ xin,
                                              const float* __restrict__ w,
                                              const float* __restrict__ bias,
                                              float* __restrict__ xt) {
  __shared__ __align__(16) float lds_a[2][64 * 32];   // [m][k] f32, TDM dest
  __shared__ _Float16 lds_b[64 * 32];                 // [s][k] (column-major B)
  const int tid = threadIdx.x, lane = tid & 31, wid = tid >> 5;
  const int b = blockIdx.z;
  const int m_base = blockIdx.x * 64;
  const int n_base = blockIdx.y * 64;
  const float* xb = xin + (size_t)b * (NC * NHW);
  const unsigned a_addr[2] = {lds_addr_of(&lds_a[0][0]), lds_addr_of(&lds_a[1][0])};

  v8f acc[4];
#pragma unroll
  for (int t = 0; t < 4; ++t) acc[t] = v8f_splat(0.f);

  // prologue: DMA first A chunk (64 rows x 32 f32, row stride 4096)
  if (wid == 0)
    tdm_load_2d(a_addr[0], w + (size_t)m_base * NE, 32, 64, NE, NE, NE, 2);

  const int ssc = tid >> 1, skc = (tid & 1) * 16;  // B staging: 16 elems/thread
  const int s = n_base + ssc;
  const int oy = s >> 4, ox = s & 15;

  for (int i = 0; i < NE / 32; ++i) {
    const int k0 = i * 32;
    // Stage B tile: fused im2col, zero padding handled branch-free (clamp+select)
    {
      _Float16* dst = lds_b + ssc * 32 + skc;
#pragma unroll
      for (int j = 0; j < 16; ++j) {
        int kk = k0 + skc + j;
        int c = kk >> 6, rem = kk & 63;
        int h  = oy * 8 + (rem >> 3) - 8;
        int ww = ox * 8 + (rem & 7) - 8;
        bool in = ((unsigned)h < 112u) && ((unsigned)ww < 112u);
        int hc = in ? h : 0, wc = in ? ww : 0;
        float v = xb[c * NHW + hc * NIMG + wc];
        dst[j] = (_Float16)(in ? v : 0.f);
      }
    }
    if (i + 1 < NE / 32)
      __builtin_prefetch(xb + (size_t)((k0 + 32) >> 6) * NHW, 0, 1);
    // DMA next A chunk into the other buffer; ensure current chunk has landed.
    if (wid == 0) {
      if (i + 1 < NE / 32) {
        tdm_load_2d(a_addr[(i + 1) & 1], w + (size_t)m_base * NE + (k0 + 32),
                    32, 64, NE, NE, NE, 2);
        __builtin_amdgcn_s_wait_tensorcnt(1);
      } else {
        __builtin_amdgcn_s_wait_tensorcnt(0);
      }
    }
    __syncthreads();

    v16h a = frag_a_f32(lds_a[i & 1] + (wid * 16) * 32, 32);
#pragma unroll
    for (int t = 0; t < 4; ++t) {
      v16h bf = frag_b_f16(lds_b + (t * 16) * 32, 32);
      acc[t] = wmma32(a, bf, acc[t]);
    }
    __syncthreads();
  }

  const int col = lane & 15, rbase = (lane >> 4) * 8;
#pragma unroll
  for (int t = 0; t < 4; ++t) {
    int sg = n_base + t * 16 + col;
#pragma unroll
    for (int r = 0; r < 8; ++r) {
      int m = m_base + wid * 16 + rbase + r;
      xt[((size_t)b * NE + m) * NS + sg] = acc[t][r] + bias[m];
    }
  }
}

// ---------------------------------------------------------------------------
// Kernel 2: sequential neighbor-average scan (exactly replicates the jax scan,
// including the pn=14 stride on the 256-long row). One thread per (b,e) row;
// row lives in LDS (padded to avoid bank conflicts). Grid 512, block 32.
// ---------------------------------------------------------------------------
__global__ __launch_bounds__(32) void k_scan(float* __restrict__ xt) {
  __shared__ float rows[32][257];
  const int t = threadIdx.x;
  const size_t g = (size_t)blockIdx.x * 32 + t;  // flat (b,e)
  float* rp = xt + g * NS;
  float* r = rows[t];
  for (int j = 0; j < NS; ++j) r[j] = rp[j];
  for (int i = 1; i <= 12; ++i)
    for (int j = 1; j <= 13; ++j) {
      int idx = i * 14 + j;
      float sum = r[idx - 14] + r[idx + 14] + r[idx - 1] + r[idx + 1] +
                  r[idx - 15] + r[idx - 13] + r[idx + 13] + r[idx + 15];
      r[idx] = sum * 0.125f;
    }
  for (int j = 0; j < NS; ++j) rp[j] = r[j];
}

// ---------------------------------------------------------------------------
// Kernel 3: reshape(4096,256)->(64,128,128), center-crop, fc1 1x1 conv (WMMA,
// M=64,K=64) + per-channel sum/sumsq for BN1. Grid (196, B), block 128.
// ---------------------------------------------------------------------------
__global__ __launch_bounds__(128) void k_fc1(const float* __restrict__ xt,
                                             const float* __restrict__ w,
                                             const float* __restrict__ bias,
                                             float* __restrict__ u,
                                             float* __restrict__ stats) {
  __shared__ _Float16 lds_b[64 * 64];  // [n][c] column-major B, ld 64
  const int tid = threadIdx.x, lane = tid & 31, wid = tid >> 5;
  const int b = blockIdx.y;
  const int n_base = blockIdx.x * 64;

  {
    int n = tid >> 1, cbase = (tid & 1) * 32;
    int ng = n_base + n;
    int base_hw = (ng / 112 + 8) * 128 + (ng % 112 + 8);
    const float* xtb = xt + (size_t)b * (NE * NS);  // flat (e*256+s) == (c*16384 + h*128 + w)
    _Float16* dst = lds_b + n * 64 + cbase;
#pragma unroll
    for (int j = 0; j < 32; ++j)
      dst[j] = (_Float16)xtb[(cbase + j) * 16384 + base_hw];
  }
  __syncthreads();

  const int o_tile = wid * 16;
  v16h a0 = frag_a_f32(w + o_tile * 64, 64);
  v16h a1 = frag_a_f32(w + o_tile * 64 + 32, 64);

  float psum[8], psq[8];
#pragma unroll
  for (int r = 0; r < 8; ++r) { psum[r] = 0.f; psq[r] = 0.f; }

  const int col = lane & 15, rbase = (lane >> 4) * 8;
#pragma unroll
  for (int t = 0; t < 4; ++t) {
    v8f acc = v8f_splat(0.f);
    v16h b0 = frag_b_f16(lds_b + (t * 16) * 64, 64);
    v16h b1 = frag_b_f16(lds_b + (t * 16) * 64 + 32, 64);
    acc = wmma32(a0, b0, acc);
    acc = wmma32(a1, b1, acc);
    int ng = n_base + t * 16 + col;
#pragma unroll
    for (int r = 0; r < 8; ++r) {
      int o = o_tile + rbase + r;
      float v = acc[r] + bias[o];
      u[((size_t)b * NC + o) * NHW + ng] = v;
      psum[r] += v;
      psq[r] += v * v;
    }
  }
#pragma unroll
  for (int r = 0; r < 8; ++r) {
    float sv = psum[r], qv = psq[r];
#pragma unroll
    for (int m = 8; m >= 1; m >>= 1) {
      sv += __shfl_xor(sv, m, 16);
      qv += __shfl_xor(qv, m, 16);
    }
    if ((lane & 15) == 0) {
      int o = o_tile + rbase + r;
      atomicAdd(&stats[o], sv);
      atomicAdd(&stats[NC + o], qv);
    }
  }
}

// ---------------------------------------------------------------------------
// Kernel 4: fold BN stats -> per-channel scale/shift
// ---------------------------------------------------------------------------
__global__ void k_bn_fin(const float* __restrict__ stats, const float* __restrict__ g,
                         const float* __restrict__ bcf, float* __restrict__ ss,
                         int count, float invN) {
  int i = threadIdx.x;
  if (i < count) {
    float mean = stats[i] * invN;
    float var = stats[count + i] * invN - mean * mean;
    float sc = g[i] * rsqrtf(var + 1e-5f);
    ss[i] = sc;
    ss[count + i] = bcf[i] - mean * sc;
  }
}

// ---------------------------------------------------------------------------
// Kernel 5: apply BN1 and transpose to [b][n][c] f16 (feeds KNN A/B and feat)
// ---------------------------------------------------------------------------
__global__ __launch_bounds__(256) void k_bn_apply_T(const float* __restrict__ u,
                                                    const float* __restrict__ ss,
                                                    _Float16* __restrict__ o16) {
  size_t g = (size_t)blockIdx.x * 256 + threadIdx.x;  // b*HW + n
  int b = (int)(g / NHW), n = (int)(g % NHW);
  const float* ub = u + (size_t)b * NC * NHW + n;
  _Float16* dst = o16 + g * NC;
#pragma unroll
  for (int c = 0; c < NC; ++c)
    dst[c] = (_Float16)(ub[(size_t)c * NHW] * ss[c] + ss[NC + c]);
}

// ---------------------------------------------------------------------------
// Kernel 6: 2x2 avg-pool of yf -> zn (f16) and candidate-side unit vectors zu (f16)
// ---------------------------------------------------------------------------
__global__ __launch_bounds__(64) void k_z(const _Float16* __restrict__ yn16,
                                          _Float16* __restrict__ zn16,
                                          _Float16* __restrict__ zu16) {
  size_t g = (size_t)blockIdx.x * 64 + threadIdx.x;  // b*NZ + m
  int b = (int)(g / NZ), m = (int)(g % NZ);
  int i = m / 56, j = m % 56;
  const _Float16* p00 = yn16 + ((size_t)b * NHW + (2 * i) * NIMG + 2 * j) * NC;
  const _Float16* p01 = p00 + NC;
  const _Float16* p10 = p00 + (size_t)NIMG * NC;
  const _Float16* p11 = p10 + NC;
  float v[NC];
  float nrm = 0.f;
#pragma unroll
  for (int c = 0; c < NC; ++c) {
    float t = 0.25f * ((float)p00[c] + (float)p01[c] + (float)p10[c] + (float)p11[c]);
    v[c] = t;
    nrm += t * t;
  }
  float rs = 1.f / fmaxf(sqrtf(nrm), 1e-12f);
  _Float16* zd = zn16 + g * NC;
  _Float16* ud = zu16 + g * NC;
#pragma unroll
  for (int c = 0; c < NC; ++c) {
    zd[c] = (_Float16)v[c];
    ud[c] = (_Float16)(v[c] * rs);
  }
}

// ---------------------------------------------------------------------------
// Kernel 7: KNN. Scores S[m][n] = zu_m . xn_n via WMMA (A = candidates, B = queries
// so each lane owns one query), fused per-lane top-9, half-wave merge in LDS.
// Query-side normalization is dropped: positive per-query scaling preserves ranking.
// Grid (196, B), block 128 (4 waves x 16 queries).
// ---------------------------------------------------------------------------
__global__ __launch_bounds__(128) void k_knn(const _Float16* __restrict__ xn16,
                                             const _Float16* __restrict__ zu16,
                                             int* __restrict__ nn) {
  __shared__ _Float16 atile[16 * 64];
  __shared__ float msco[128 * 9];
  __shared__ int midx[128 * 9];
  const int tid = threadIdx.x, lane = tid & 31, wid = tid >> 5;
  const int b = blockIdx.y;
  const int q_base = blockIdx.x * 64 + wid * 16;
  const int col = lane & 15;

  // Persistent query fragments (K=64 -> 2 chunks of 32)
  const _Float16* qp =
      xn16 + ((size_t)b * NHW + q_base + col) * NC + ((lane >> 4) * 16);
  v16h bq0, bq1;
#pragma unroll
  for (int j = 0; j < 16; ++j) { bq0[j] = qp[j]; bq1[j] = qp[32 + j]; }

  float ts[NK];
  int ti[NK];
#pragma unroll
  for (int j = 0; j < NK; ++j) { ts[j] = -1e30f; ti[j] = 0; }

  const int ldr = tid >> 3, ldo = (tid & 7) * 8;
  for (int mt = 0; mt < NZ / 16; ++mt) {
    int m0 = mt * 16;
    {
      const _Float16* s = zu16 + ((size_t)b * NZ + m0 + ldr) * NC + ldo;
      _Float16* d = atile + ldr * 64 + ldo;
#pragma unroll
      for (int j = 0; j < 8; ++j) d[j] = s[j];
    }
    __syncthreads();
    v16h a0 = frag_a_f16(atile, 64);
    v16h a1 = frag_a_f16(atile + 32, 64);
    v8f acc = v8f_splat(0.f);
    acc = wmma32(a0, bq0, acc);
    acc = wmma32(a1, bq1, acc);
    int mb = m0 + (lane >> 4) * 8;
#pragma unroll
    for (int r = 0; r < 8; ++r) {
      float v = acc[r];
      int m = mb + r;
      float mv = ts[0];
      int mp = 0;
#pragma unroll
      for (int j = 1; j < NK; ++j)
        if (ts[j] < mv) { mv = ts[j]; mp = j; }
      bool take = v > mv;
#pragma unroll
      for (int j = 0; j < NK; ++j) {
        bool sel = take && (j == mp);
        ts[j] = sel ? v : ts[j];
        ti[j] = sel ? m : ti[j];
      }
    }
    __syncthreads();
  }

#pragma unroll
  for (int j = 0; j < NK; ++j) { msco[tid * NK + j] = ts[j]; midx[tid * NK + j] = ti[j]; }
  __syncthreads();
  if (lane < 16) {
    int base0 = (wid * 32 + lane) * NK;
    int base1 = (wid * 32 + lane + 16) * NK;
    int* dst = nn + ((size_t)b * NHW + q_base + lane) * NK;
    unsigned used = 0;
    for (int k = 0; k < NK; ++k) {
      float best = -1e30f;
      int bj = 0;
      for (int j = 0; j < 18; ++j) {
        if (used & (1u << j)) continue;
        float v = (j < NK) ? msco[base0 + j] : msco[base1 + j - NK];
        if (v > best) { best = v; bj = j; }
      }
      used |= 1u << bj;
      dst[k] = (bj < NK) ? midx[base0 + bj] : midx[base1 + bj - NK];
    }
  }
}

// ---------------------------------------------------------------------------
// Kernel 8: gather feat = [x_i ; x_j - x_i], gc GEMM (128x128) via WMMA, then
// elementwise max over the 9 neighbor accumulators (relu.max == relu(max)).
// Grid (784, B), block 256 (8 waves x 16 output rows), 16 positions per WG.
// ---------------------------------------------------------------------------
__global__ __launch_bounds__(256) void k_gc(const _Float16* __restrict__ xn16,
                                            const _Float16* __restrict__ zn16,
                                            const int* __restrict__ nn,
                                            const float* __restrict__ gw,
                                            const float* __restrict__ gb,
                                            float* __restrict__ hmax) {
  __shared__ _Float16 bfeat[16 * 128];  // [n][c], ld 128
  const int tid = threadIdx.x, lane = tid & 31, wid = tid >> 5;
  const int b = blockIdx.y;
  const int n0 = blockIdx.x * 16;
  const int o_tile = wid * 16;

  v16h af[4];
#pragma unroll
  for (int ch = 0; ch < 4; ++ch)
    af[ch] = frag_a_f32(gw + o_tile * 128 + ch * 32, 128);

  const int bn = tid & 15, cbase = (tid >> 4) * 8;  // 16 n x 16 c-groups of 8
  const int ng = n0 + bn;
  const _Float16* xrow = xn16 + ((size_t)b * NHW + ng) * NC;
  const int* nrow = nn + ((size_t)b * NHW + ng) * NK;

  v8f mxa = v8f_splat(-1e30f);
  for (int k = 0; k < NK; ++k) {
    _Float16* d = bfeat + bn * 128 + cbase;
    if (cbase < NC) {
#pragma unroll
      for (int j = 0; j < 8; ++j) d[j] = xrow[cbase + j];
    } else {
      int idx = nrow[k];
      const _Float16* zrow = zn16 + ((size_t)b * NZ + idx) * NC + (cbase - NC);
      const _Float16* xr2 = xrow + (cbase - NC);
#pragma unroll
      for (int j = 0; j < 8; ++j)
        d[j] = (_Float16)((float)zrow[j] - (float)xr2[j]);
    }
    __syncthreads();
    v8f acc = v8f_splat(0.f);
#pragma unroll
    for (int ch = 0; ch < 4; ++ch) {
      v16h bf = frag_b_f16(bfeat + ch * 32, 128);
      acc = wmma32(af[ch], bf, acc);
    }
#pragma unroll
    for (int r = 0; r < 8; ++r) mxa[r] = fmaxf(mxa[r], acc[r]);
    __syncthreads();
  }

  const int col = lane & 15, rbase = (lane >> 4) * 8;
#pragma unroll
  for (int r = 0; r < 8; ++r) {
    int o = o_tile + rbase + r;
    hmax[((size_t)b * 128 + o) * NHW + (n0 + col)] = fmaxf(mxa[r] + gb[o], 0.f);
  }
}

// ---------------------------------------------------------------------------
// Kernel 9: fc2 (64x128) via WMMA + BN2 stats. B tile (128 x 16 f32, row
// stride 12544) is DMA'd to LDS by the Tensor Data Mover. Grid (784, B), block 128.
// ---------------------------------------------------------------------------
__global__ __launch_bounds__(128) void k_fc2(const float* __restrict__ hmax,
                                             const float* __restrict__ w,
                                             const float* __restrict__ bias,
                                             float* __restrict__ u2,
                                             float* __restrict__ stats) {
  __shared__ __align__(16) float bh32[128 * 16];  // [c][n] f32, TDM dest
  const int tid = threadIdx.x, lane = tid & 31, wid = tid >> 5;
  const int b = blockIdx.y;
  const int n0 = blockIdx.x * 16;

  if (wid == 0) {
    tdm_load_2d(lds_addr_of(bh32), hmax + (size_t)b * 128 * NHW + n0,
                16, 128, NHW, 128, NHW, 2);
    __builtin_amdgcn_s_wait_tensorcnt(0);
  }
  __syncthreads();

  const int o_tile = wid * 16;
  v8f acc = v8f_splat(0.f);
#pragma unroll
  for (int ch = 0; ch < 4; ++ch) {
    v16h a = frag_a_f32(w + o_tile * 128 + ch * 32, 128);
    v16h bf = frag_b_f32k(bh32 + ch * 32 * 16, 16);
    acc = wmma32(a, bf, acc);
  }
  const int col = lane & 15, rbase = (lane >> 4) * 8;
#pragma unroll
  for (int r = 0; r < 8; ++r) {
    int o = o_tile + rbase + r;
    float v = acc[r] + bias[o];
    u2[((size_t)b * NC + o) * NHW + n0 + col] = v;
    float sv = v, qv = v * v;
#pragma unroll
    for (int m = 8; m >= 1; m >>= 1) {
      sv += __shfl_xor(sv, m, 16);
      qv += __shfl_xor(qv, m, 16);
    }
    if (col == 0) {
      atomicAdd(&stats[o], sv);
      atomicAdd(&stats[NC + o], qv);
    }
  }
}

// ---------------------------------------------------------------------------
// Kernel 10: apply BN2, add residuals, write both tuple outputs.
// ---------------------------------------------------------------------------
__global__ __launch_bounds__(256) void k_final(const float* __restrict__ u2,
                                               const float* __restrict__ ss,
                                               const float* __restrict__ x,
                                               const float* __restrict__ y,
                                               float* __restrict__ out) {
  size_t i = (size_t)blockIdx.x * 256 + threadIdx.x;
  int c = (int)((i / NHW) & 63);
  float v = u2[i] * ss[c] + ss[NC + c];
  out[i] = v + x[i];
  out[(size_t)NTOT + i] = v + y[i];
}

// ---------------------------------------------------------------------------
// Launcher
// ---------------------------------------------------------------------------
extern "C" void kernel_launch(void* const* d_in, const int* in_sizes, int n_in,
                              void* d_out, int out_size, void* d_ws, size_t ws_size,
                              hipStream_t stream) {
  (void)in_sizes; (void)n_in; (void)out_size; (void)ws_size;
  const float* x      = (const float*)d_in[0];
  const float* y      = (const float*)d_in[1];
  const float* cagg_w = (const float*)d_in[2];
  const float* cagg_b = (const float*)d_in[3];
  const float* fc1_w  = (const float*)d_in[4];
  const float* fc1_b  = (const float*)d_in[5];
  const float* bn1_g  = (const float*)d_in[6];
  const float* bn1_b  = (const float*)d_in[7];
  const float* gc_w   = (const float*)d_in[8];
  const float* gc_b   = (const float*)d_in[9];
  const float* fc2_w  = (const float*)d_in[10];
  const float* fc2_b  = (const float*)d_in[11];
  const float* bn2_g  = (const float*)d_in[12];
  const float* bn2_b  = (const float*)d_in[13];
  float* out = (float*)d_out;

  char* ws = (char*)d_ws;
  size_t off = 0;
  auto alloc = [&](size_t bytes) -> char* {
    char* p = ws + off;
    off += (bytes + 255) & ~(size_t)255;
    return p;
  };
  float*     xt_x   = (float*)alloc((size_t)NB * NE * NS * 4);
  float*     xt_y   = (float*)alloc((size_t)NB * NE * NS * 4);
  float*     u_x    = (float*)alloc((size_t)NB * NC * NHW * 4);
  float*     u_y    = (float*)alloc((size_t)NB * NC * NHW * 4);
  _Float16*  xn16   = (_Float16*)alloc((size_t)NB * NHW * NC * 2);
  _Float16*  yn16   = (_Float16*)alloc((size_t)NB * NHW * NC * 2);
  _Float16*  zn16   = (_Float16*)alloc((size_t)NB * NZ * NC * 2);
  _Float16*  zu16   = (_Float16*)alloc((size_t)NB * NZ * NC * 2);
  int*       nn     = (int*)alloc((size_t)NB * NHW * NK * 4);
  float*     hmax   = (float*)alloc((size_t)NB * 128 * NHW * 4);
  float*     stats  = (float*)alloc(3 * 128 * 4);  // [bn1x | bn1y | bn2] sum+sumsq
  float*     ss1x   = (float*)alloc(128 * 4);
  float*     ss1y   = (float*)alloc(128 * 4);
  float*     ss2    = (float*)alloc(128 * 4);
  float*     u2     = u_x;  // u_x is dead after k_bn_apply_T(x); reuse for fc2 output

  const float invN = 1.0f / (float)(NB * NHW);

  k_zero<<<dim3(2), dim3(256), 0, stream>>>(stats, 3 * 128);

  k_cagg<<<dim3(64, 4, NB), dim3(128), 0, stream>>>(x, cagg_w, cagg_b, xt_x);
  k_cagg<<<dim3(64, 4, NB), dim3(128), 0, stream>>>(y, cagg_w, cagg_b, xt_y);

  k_scan<<<dim3(512), dim3(32), 0, stream>>>(xt_x);
  k_scan<<<dim3(512), dim3(32), 0, stream>>>(xt_y);

  k_fc1<<<dim3(196, NB), dim3(128), 0, stream>>>(xt_x, fc1_w, fc1_b, u_x, stats);
  k_fc1<<<dim3(196, NB), dim3(128), 0, stream>>>(xt_y, fc1_w, fc1_b, u_y, stats + 128);

  k_bn_fin<<<dim3(1), dim3(64), 0, stream>>>(stats, bn1_g, bn1_b, ss1x, NC, invN);
  k_bn_fin<<<dim3(1), dim3(64), 0, stream>>>(stats + 128, bn1_g, bn1_b, ss1y, NC, invN);

  k_bn_apply_T<<<dim3(196), dim3(256), 0, stream>>>(u_x, ss1x, xn16);
  k_bn_apply_T<<<dim3(196), dim3(256), 0, stream>>>(u_y, ss1y, yn16);

  k_z<<<dim3(196), dim3(64), 0, stream>>>(yn16, zn16, zu16);

  k_knn<<<dim3(196, NB), dim3(128), 0, stream>>>(xn16, zu16, nn);

  k_gc<<<dim3(784, NB), dim3(256), 0, stream>>>(xn16, zn16, nn, gc_w, gc_b, hmax);

  k_fc2<<<dim3(784, NB), dim3(128), 0, stream>>>(hmax, fc2_w, fc2_b, u2, stats + 256);

  k_bn_fin<<<dim3(1), dim3(64), 0, stream>>>(stats + 256, bn2_g, bn2_b, ss2, NC, invN);

  k_final<<<dim3(12544), dim3(256), 0, stream>>>(u2, ss2, x, y, out);
}
